// Decoder_48747878809925
// MI455X (gfx1250) — matrix operand order repaired and checked by reading, not compiled, not run on hardware
//
#include <hip/hip_runtime.h>
#include <hip/hip_bf16.h>
#include <math.h>

#define VOCAB  32000
#define DMODEL 1024
#define HID    1024
#define NLAYERS 4
#define BATCH  32
#define SEQ    128
#define G4H    4096   // 4*HID
#define NTOK   4096   // BATCH*SEQ

typedef __attribute__((ext_vector_type(16))) __bf16 v16bf;
typedef __attribute__((ext_vector_type(8)))  __bf16 v8bf;
typedef __attribute__((ext_vector_type(8)))  float  v8f;

__device__ __forceinline__ unsigned short f32_to_bf16_rne(float f) {
  union { float f; unsigned int u; } v; v.f = f;
  unsigned int u = v.u;
  u += 0x7FFFu + ((u >> 16) & 1u);   // round-to-nearest-even
  return (unsigned short)(u >> 16);
}

__device__ __forceinline__ v8f wmma_bf16(v16bf a, v16bf b, v8f c) {
  // (neg_a, A, neg_b, B, c_mod, C, reuse_a, reuse_b)
  return __builtin_amdgcn_wmma_f32_16x16x32_bf16(false, a, false, b, (short)0, c,
                                                 false, false);
}

// A fragment (16x32 bf16): lanes 0-15 row=l, elems 0..15 = K+0..15
//                          lanes16-31 row=l, elems 0..15 = K+16..31
__device__ __forceinline__ v16bf load_a_frag(const unsigned short* row, int k, int hi) {
  return *(const v16bf*)(row + k + hi * 16);
}

__device__ __forceinline__ float sigmoidf_(float x) { return 1.0f / (1.0f + __expf(-x)); }

// ---------------------------------------------------------------- conversions
__global__ void __launch_bounds__(256) cvt_f32_bf16_kernel(
    const float* __restrict__ in, unsigned short* __restrict__ out, long n) {
  long i = (long)blockIdx.x * 256 + threadIdx.x;
  long stride = (long)gridDim.x * 256;
  for (; i < n; i += stride) out[i] = f32_to_bf16_rne(in[i]);
}

// ------------------------------------------------- embedding + pos-encoding
__global__ void __launch_bounds__(256) embed_pe_kernel(
    const int* __restrict__ tgt, const float* __restrict__ emb,
    unsigned short* __restrict__ out) {
  int token = blockIdx.x;                 // b*SEQ + s
  int s = token & (SEQ - 1);
  int id = tgt[token];
  const float* erow = emb + (size_t)id * DMODEL;
  const float cexp = -9.210340371976184f / (float)DMODEL;  // -ln(10000)/D
  for (int d = threadIdx.x; d < DMODEL; d += 256) {
    float freq = expf((float)(d & ~1) * cexp);
    float ang = (float)s * freq;
    float pe = (d & 1) ? cosf(ang) : sinf(ang);
    out[(size_t)token * DMODEL + d] = f32_to_bf16_rne(erow[d] + pe);
  }
}

// --------------------------------------------------------- bf16 WMMA GEMM
// out[M x N] = A[M x K] * W[N x K]^T + bias0[n] + bias1[n]
// block = 4 waves (128 thr); block tile 64(M) x 128(N); wave tile 32 x 64.
__global__ void __launch_bounds__(128) gemm_bf16_kernel(
    const unsigned short* __restrict__ A, const unsigned short* __restrict__ W,
    const float* __restrict__ bias0, const float* __restrict__ bias1,
    float* __restrict__ out, int N, int K) {
  const int lane = threadIdx.x & 31;
  const int wave = threadIdx.x >> 5;
  const int l  = lane & 15;
  const int hi = lane >> 4;
  const int wm = wave & 1;
  const int wn = wave >> 1;
  const int m0 = blockIdx.y * 64 + wm * 32;
  const int n0 = blockIdx.x * 128 + wn * 64;

  v8f acc[2][4];
#pragma unroll
  for (int mt = 0; mt < 2; mt++)
#pragma unroll
    for (int nt = 0; nt < 4; nt++) acc[mt][nt] = (v8f)(0.0f);

  const unsigned short* Ar0 = A + (size_t)(m0 + l) * K;
  const unsigned short* Ar1 = A + (size_t)(m0 + 16 + l) * K;
  const unsigned short* Wr[4];
#pragma unroll
  for (int nt = 0; nt < 4; nt++)
    Wr[nt] = W + (size_t)(n0 + nt * 16 + l) * K + hi * 16;

#pragma unroll 2
  for (int k = 0; k < K; k += 32) {
    __builtin_prefetch(Wr[0] + k + 512, 0, 0);  // global_prefetch_b8
    v16bf a0 = load_a_frag(Ar0, k, hi);
    v16bf a1 = load_a_frag(Ar1, k, hi);
#pragma unroll
    for (int nt = 0; nt < 4; nt++) {
      v16bf b = *(const v16bf*)(Wr[nt] + k);   // B frag: lane l = col, hi half = K+16
      acc[0][nt] = wmma_bf16(a0, b, acc[0][nt]);
      acc[1][nt] = wmma_bf16(a1, b, acc[1][nt]);
    }
  }
  // D layout: elem r, lane (l,hi) -> M = r + 8*hi, N = l
#pragma unroll
  for (int mt = 0; mt < 2; mt++)
#pragma unroll
    for (int nt = 0; nt < 4; nt++)
#pragma unroll
      for (int r = 0; r < 8; r++) {
        int m = m0 + mt * 16 + r + 8 * hi;
        int n = n0 + nt * 16 + l;
        float v = acc[mt][nt][r];
        if (bias0) v += bias0[n];
        if (bias1) v += bias1[n];
        out[(size_t)m * N + n] = v;
      }
}

// ------------------------------------------------------------- grid barrier
__device__ __forceinline__ void grid_barrier(unsigned int* cnt, unsigned int target) {
  __threadfence();
  __syncthreads();
  if (threadIdx.x == 0) {
    __hip_atomic_fetch_add(cnt, 1u, __ATOMIC_ACQ_REL, __HIP_MEMORY_SCOPE_AGENT);
    while (__hip_atomic_load(cnt, __ATOMIC_ACQUIRE, __HIP_MEMORY_SCOPE_AGENT) < target)
      __builtin_amdgcn_s_sleep(4);
  }
  __syncthreads();
}

// -------------------------------------------------------- persistent LSTM scan
// 32 blocks x 128 threads. Block b owns hidden units [b*32, b*32+32), all 4 gates.
// Wave (wm,wu): batch tile wm*16, unit tile wu*16; holds i/f/g/o + c in registers.
__global__ void __launch_bounds__(128) lstm_scan_kernel(
    const float* __restrict__ xg,            // [NTOK x 4096], row = m*SEQ + t
    const unsigned short* __restrict__ whh,  // [4096 x 1024] bf16
    const float* __restrict__ h0l, const float* __restrict__ c0l,  // [32 x 1024]
    unsigned short* __restrict__ hbuf,       // [2 x 32 x 1024] bf16 (double buffer)
    unsigned short* __restrict__ y,          // [NTOK x 1024] bf16
    float* __restrict__ hn_out, float* __restrict__ cn_out,        // [32 x 1024]
    unsigned int* __restrict__ counter) {
  const int lane = threadIdx.x & 31;
  const int wave = threadIdx.x >> 5;
  const int l  = lane & 15;
  const int hi = lane >> 4;
  const int wm = wave & 1;
  const int wu = wave >> 1;
  const int mb = wm * 16;
  const int ub = blockIdx.x * 32 + wu * 16;
  const unsigned int nb = gridDim.x;

  // init h double-buffer slot 0 (this block's 32 units, all 32 batch rows)
  for (int idx = threadIdx.x; idx < 32 * 32; idx += 128) {
    int mm = idx >> 5;
    int uu = (idx & 31) + blockIdx.x * 32;
    hbuf[mm * HID + uu] = f32_to_bf16_rne(h0l[mm * HID + uu]);
  }
  // c tile kept in registers across all timesteps (D layout)
  v8f c;
#pragma unroll
  for (int r = 0; r < 8; r++) c[r] = c0l[(mb + r + 8 * hi) * HID + ub + l];

  grid_barrier(counter, nb);

  const unsigned short* wrow[4];
#pragma unroll
  for (int gg = 0; gg < 4; gg++)
    wrow[gg] = whh + (size_t)(gg * 1024 + ub + l) * HID + hi * 16;

  for (int t = 0; t < SEQ; t++) {
    const unsigned short* hread = hbuf + (t & 1) * (32 * HID);
    unsigned short* hwrite      = hbuf + ((t + 1) & 1) * (32 * HID);

    // accumulators seeded with precomputed input-gate preactivations
    v8f g[4];
#pragma unroll
    for (int gg = 0; gg < 4; gg++) {
      const float* xgp = xg + (size_t)t * G4H + gg * 1024 + ub + l;
#pragma unroll
      for (int r = 0; r < 8; r++) {
        int m = mb + r + 8 * hi;
        g[gg][r] = xgp[(size_t)m * SEQ * G4H];
      }
    }

    const unsigned short* hrow = hread + (mb + l) * HID;
#pragma unroll 4
    for (int k = 0; k < HID; k += 32) {
      v16bf a = load_a_frag(hrow, k, hi);
#pragma unroll
      for (int gg = 0; gg < 4; gg++) {
        v16bf b = *(const v16bf*)(wrow[gg] + k);
        g[gg] = wmma_bf16(a, b, g[gg]);
      }
    }

    v8f hnew;
#pragma unroll
    for (int r = 0; r < 8; r++) {
      float ig = sigmoidf_(g[0][r]);
      float fg = sigmoidf_(g[1][r]);
      float gc = tanhf(g[2][r]);
      float og = sigmoidf_(g[3][r]);
      float cc = fg * c[r] + ig * gc;
      c[r] = cc;
      hnew[r] = og * tanhf(cc);
    }
#pragma unroll
    for (int r = 0; r < 8; r++) {
      int m = mb + r + 8 * hi;
      unsigned short hb = f32_to_bf16_rne(hnew[r]);
      hwrite[m * HID + ub + l] = hb;
      y[((size_t)m * SEQ + t) * HID + ub + l] = hb;
    }
    if (t == SEQ - 1) {
#pragma unroll
      for (int r = 0; r < 8; r++) {
        int m = mb + r + 8 * hi;
        hn_out[m * HID + ub + l] = hnew[r];
        cn_out[m * HID + ub + l] = c[r];
      }
    }
    grid_barrier(counter, (unsigned int)(t + 2) * nb);
  }
}

// ---------------------------------------------------------------- launcher
extern "C" void kernel_launch(void* const* d_in, const int* in_sizes, int n_in,
                              void* d_out, int out_size, void* d_ws, size_t ws_size,
                              hipStream_t stream) {
  (void)in_sizes; (void)n_in; (void)out_size; (void)ws_size;
  const int*   tgt   = (const int*)d_in[0];
  const float* h0    = (const float*)d_in[1];
  const float* c0    = (const float*)d_in[2];
  const float* emb   = (const float*)d_in[3];
  const float* w_ih  = (const float*)d_in[4];
  const float* w_hh  = (const float*)d_in[5];
  const float* b_ih  = (const float*)d_in[6];
  const float* b_hh  = (const float*)d_in[7];
  const float* w_cls = (const float*)d_in[8];
  const float* b_cls = (const float*)d_in[9];
  float* out = (float*)d_out;

  char* ws = (char*)d_ws;
  size_t off = 0;
  auto carve = [&](size_t bytes) -> char* {
    char* p = ws + off;
    off = (off + bytes + 255) & ~(size_t)255;
    return p;
  };
  unsigned short* wihb  = (unsigned short*)carve((size_t)NLAYERS * G4H * DMODEL * 2);
  unsigned short* whhb  = (unsigned short*)carve((size_t)NLAYERS * G4H * HID * 2);
  unsigned short* wclsb = (unsigned short*)carve((size_t)VOCAB * HID * 2);
  float*          xg    = (float*)carve((size_t)NTOK * G4H * 4);
  unsigned short* bufA  = (unsigned short*)carve((size_t)NTOK * HID * 2);
  unsigned short* bufB  = (unsigned short*)carve((size_t)NTOK * HID * 2);
  unsigned short* hbuf  = (unsigned short*)carve((size_t)2 * BATCH * HID * 2);
  unsigned int*   cnt   = (unsigned int*)carve(256);

  cvt_f32_bf16_kernel<<<2048, 256, 0, stream>>>(w_ih, wihb, (long)NLAYERS * G4H * DMODEL);
  cvt_f32_bf16_kernel<<<2048, 256, 0, stream>>>(w_hh, whhb, (long)NLAYERS * G4H * HID);
  cvt_f32_bf16_kernel<<<2048, 256, 0, stream>>>(w_cls, wclsb, (long)VOCAB * HID);
  embed_pe_kernel<<<NTOK, 256, 0, stream>>>(tgt, emb, bufA);

  unsigned short* x = bufA;
  unsigned short* yb = bufB;
  const size_t hn_off = (size_t)NTOK * VOCAB;                    // logits elems
  const size_t cn_off = hn_off + (size_t)NLAYERS * BATCH * HID;

  for (int lyr = 0; lyr < NLAYERS; ++lyr) {
    dim3 g1(G4H / 128, NTOK / 64);
    gemm_bf16_kernel<<<g1, 128, 0, stream>>>(
        x, wihb + (size_t)lyr * G4H * DMODEL,
        b_ih + (size_t)lyr * G4H, b_hh + (size_t)lyr * G4H, xg, G4H, DMODEL);
    hipMemsetAsync(cnt, 0, 256, stream);
    lstm_scan_kernel<<<32, 128, 0, stream>>>(
        xg, whhb + (size_t)lyr * G4H * HID,
        h0 + (size_t)lyr * BATCH * HID, c0 + (size_t)lyr * BATCH * HID,
        hbuf, yb,
        out + hn_off + (size_t)lyr * BATCH * HID,
        out + cn_off + (size_t)lyr * BATCH * HID, cnt);
    unsigned short* tmp = x; x = yb; yb = tmp;
  }
  dim3 g2(VOCAB / 128, NTOK / 64);
  gemm_bf16_kernel<<<g2, 128, 0, stream>>>(x, wclsb, b_cls, nullptr, out, VOCAB, DMODEL);
}
